// KBBlock_s_80015240724678
// MI455X (gfx1250) — matrix operand order repaired and checked by reading, not compiled, not run on hardware
//
#include <hip/hip_runtime.h>
#include <cstdint>
#include <cstddef>

typedef __attribute__((ext_vector_type(16))) _Float16 v16h;
typedef __attribute__((ext_vector_type(8)))  _Float16 v8h;
typedef __attribute__((ext_vector_type(8)))  float    v8f;
typedef __attribute__((ext_vector_type(2)))  float    v2f;

#define Hd 160
#define Wd 160
#define HWp 25600
#define Bn 2
#define Cn 64

// ---------------------------------------------------------------------------
// Pointwise / utility kernels
// ---------------------------------------------------------------------------
__global__ void pack_f16_kernel(const float* __restrict__ in, _Float16* __restrict__ out, int n) {
  int i = blockIdx.x * 256 + threadIdx.x;
  if (i < n) out[i] = (_Float16)in[i];
}

// f32 [B][Cc][HWp] -> f16 [B][HWp][Cc]  (N-major operand for WMMA B fragments)
__global__ void packT_f16_kernel(const float* __restrict__ in, _Float16* __restrict__ out, int Cc) {
  int t = blockIdx.x * 256 + threadIdx.x;
  if (t >= Bn * HWp) return;
  int b = t / HWp, p = t % HWp;
  const float* ib = in + (size_t)b * Cc * HWp + p;
  _Float16* ob = out + (size_t)t * Cc;
  for (int c = 0; c < Cc; c += 8) {
    v8h v;
#pragma unroll
    for (int j = 0; j < 8; ++j) v[j] = (_Float16)ib[(size_t)(c + j) * HWp];
    *(v8h*)(ob + c) = v;
  }
}

// kb_w f32 [32][2304] -> f16 [2304][32] (K-contiguous per column)
__global__ void packT_kbw_kernel(const float* __restrict__ in, _Float16* __restrict__ out) {
  int i = blockIdx.x * 256 + threadIdx.x;
  if (i >= 2304 * 32) return;
  int col = i >> 5, k = i & 31;
  out[i] = (_Float16)in[k * 2304 + col];
}

__global__ void init_dft_kernel(float* __restrict__ Fr, float* __restrict__ Fi) {
  int i = blockIdx.x * 256 + threadIdx.x;
  if (i >= Hd * Hd) return;
  int w = i / Hd, k = i % Hd;
  float ang = -2.0f * 3.14159265358979323846f * (float)((w * k) % Hd) / (float)Hd;
  float s = 0.079056941504209483f;  // 1/sqrt(160), ortho norm
  Fr[i] = cosf(ang) * s;
  Fi[i] = sinf(ang) * s;
}

__global__ void ln2d_kernel(const float* __restrict__ x, const float* __restrict__ w,
                            const float* __restrict__ bb, float* __restrict__ o) {
  int p = blockIdx.x * 256 + threadIdx.x;
  if (p >= Bn * HWp) return;
  int b = p / HWp, pp = p % HWp;
  const float* xb = x + (size_t)b * Cn * HWp + pp;
  float s = 0.f, s2 = 0.f;
  for (int c = 0; c < Cn; ++c) { float v = xb[(size_t)c * HWp]; s += v; s2 += v * v; }
  float mu = s * (1.0f / Cn);
  float var = s2 * (1.0f / Cn) - mu * mu;
  float inv = rsqrtf(var + 1e-6f);
  float* ob = o + (size_t)b * Cn * HWp + pp;
  for (int c = 0; c < Cn; ++c)
    ob[(size_t)c * HWp] = (xb[(size_t)c * HWp] - mu) * inv * w[c] + bb[c];
}

__global__ void mean_hw_kernel(const float* __restrict__ x, float* __restrict__ out) {
  int bc = blockIdx.x;
  const float* p = x + (size_t)bc * HWp;
  float s = 0.f;
  for (int i = threadIdx.x; i < HWp; i += 256) s += p[i];
  __shared__ float red[256];
  red[threadIdx.x] = s;
  __syncthreads();
  for (int st = 128; st > 0; st >>= 1) {
    if (threadIdx.x < st) red[threadIdx.x] += red[threadIdx.x + st];
    __syncthreads();
  }
  if (threadIdx.x == 0) out[bc] = red[0] * (1.0f / HWp);
}

__global__ void small_fc_kernel(const float* __restrict__ mean, int meanStride,
                                const float* __restrict__ Wt, const float* __restrict__ bias,
                                float* __restrict__ out, int Cout, int Cin, int addOne) {
  int i = blockIdx.x * 128 + threadIdx.x;
  if (i >= Bn * Cout) return;
  int b = i / Cout, c = i % Cout;
  float s = bias[c];
  for (int k = 0; k < Cin; ++k) s += Wt[c * Cin + k] * mean[b * meanStride + k];
  out[i] = addOne ? 1.0f + s : s;
}

__global__ void gconv5x5_kernel(const float* __restrict__ in, const float* __restrict__ w,
                                const float* __restrict__ bias, float* __restrict__ out) {
  int i = blockIdx.x * 256 + threadIdx.x;
  if (i >= Bn * Cn * HWp) return;
  int p = i % HWp; int co = (i / HWp) % Cn; int b = i / (Cn * HWp);
  int h = p / Wd, x = p % Wd;
  int g = co >> 2;
  float s = bias[co];
  for (int ci = 0; ci < 4; ++ci) {
    const float* ip = in + (size_t)(b * Cn + g * 4 + ci) * HWp;
    const float* wp = w + ((size_t)co * 4 + ci) * 25;
    for (int ki = 0; ki < 5; ++ki) {
      int hh = h + ki - 2;
      if (hh < 0 || hh >= Hd) continue;
      for (int kj = 0; kj < 5; ++kj) {
        int ww = x + kj - 2;
        if (ww < 0 || ww >= Wd) continue;
        s += wp[ki * 5 + kj] * ip[hh * Wd + ww];
      }
    }
  }
  out[i] = s;
}

__global__ void dwconv3x3_kernel(const float* __restrict__ in, const float* __restrict__ w,
                                 const float* __restrict__ bias, float* __restrict__ out) {
  int i = blockIdx.x * 256 + threadIdx.x;
  if (i >= Bn * Cn * HWp) return;
  int p = i % HWp; int c = (i / HWp) % Cn; int b = i / (Cn * HWp);
  int h = p / Wd, x = p % Wd;
  const float* ip = in + (size_t)(b * Cn + c) * HWp;
  const float* wp = w + (size_t)c * 9;
  float s = bias[c];
  for (int ki = 0; ki < 3; ++ki) {
    int hh = h + ki - 1;
    if (hh < 0 || hh >= Hd) continue;
    for (int kj = 0; kj < 3; ++kj) {
      int ww = x + kj - 1;
      if (ww < 0 || ww >= Wd) continue;
      s += wp[ki * 3 + kj] * ip[hh * Wd + ww];
    }
  }
  out[i] = s;
}

__global__ void gconv3x3_g32_kernel(const float* __restrict__ in, const float* __restrict__ w,
                                    const float* __restrict__ bias, float* __restrict__ out) {
  int i = blockIdx.x * 256 + threadIdx.x;
  if (i >= Bn * 32 * HWp) return;
  int p = i % HWp; int g = (i / HWp) % 32; int b = i / (32 * HWp);
  int h = p / Wd, x = p % Wd;
  float s = bias[g];
  for (int ci = 0; ci < 2; ++ci) {
    const float* ip = in + (size_t)(b * Cn + 2 * g + ci) * HWp;
    const float* wp = w + ((size_t)g * 2 + ci) * 9;
    for (int ki = 0; ki < 3; ++ki) {
      int hh = h + ki - 1;
      if (hh < 0 || hh >= Hd) continue;
      for (int kj = 0; kj < 3; ++kj) {
        int ww = x + kj - 1;
        if (ww < 0 || ww >= Wd) continue;
        s += wp[ki * 3 + kj] * ip[hh * Wd + ww];
      }
    }
  }
  out[i] = s;
}

// out[b][c][p] = in[b][c][p] * in[b][Ch+c][p], in has 2*Ch channels
__global__ void gate_kernel(const float* __restrict__ in, float* __restrict__ out, int Ch) {
  int i = blockIdx.x * 256 + threadIdx.x;
  if (i >= Bn * Ch * HWp) return;
  int p = i % HWp; int c = (i / HWp) % Ch; int b = i / (Ch * HWp);
  const float* ib = in + (size_t)b * 2 * Ch * HWp;
  out[i] = ib[(size_t)c * HWp + p] * ib[(size_t)(Ch + c) * HWp + p];
}

// att = c2c(gated)*attgamma + c211out ; emit transposed f16 [B][HW][32] (WMMA A operand)
__global__ void att_combine_kernel(const float* __restrict__ g16, const float* __restrict__ c2cw,
                                   const float* __restrict__ c2cb, const float* __restrict__ attg,
                                   const float* __restrict__ c211o, _Float16* __restrict__ attT) {
  int t = blockIdx.x * 256 + threadIdx.x;
  if (t >= Bn * HWp * 32) return;
  int s = t & 31;
  int bp = t >> 5;
  int b = bp / HWp, p = bp % HWp;
  float sum = c2cb[s];
  for (int k = 0; k < 16; ++k)
    sum += c2cw[s * 16 + k] * g16[(size_t)(b * 16 + k) * HWp + p];
  float v = sum * attg[s] + c211o[(size_t)(b * 32 + s) * HWp + p];
  attT[t] = (_Float16)v;
}

__global__ void mul3_kernel(const float* __restrict__ xk, const float* __restrict__ x1,
                            const float* __restrict__ sca, float* __restrict__ out) {
  int i = blockIdx.x * 256 + threadIdx.x;
  if (i >= Bn * Cn * HWp) return;
  int bc = i / HWp;
  out[i] = xk[i] * x1[i] * sca[bc];
}

__global__ void residual_kernel(const float* __restrict__ inp, const float* __restrict__ xc,
                                const float* __restrict__ beta, float* __restrict__ y) {
  int i = blockIdx.x * 256 + threadIdx.x;
  if (i >= Bn * Cn * HWp) return;
  int c = (i / HWp) % Cn;
  y[i] = inp[i] + xc[i] * beta[c];
}

__global__ void scale_fn_kernel(const float* __restrict__ t, const float* __restrict__ fac,
                                float* __restrict__ outR, float* __restrict__ outI) {
  int i = blockIdx.x * 256 + threadIdx.x;
  if (i >= Bn * Cn * HWp) return;
  int p = i % HWp; int c = (i / HWp) % Cn; int b = i / (Cn * HWp);
  const float* tb = t + (size_t)b * 128 * HWp;
  outR[i] = tb[(size_t)c * HWp + p] * fac[b * 64 + c];
  outI[i] = tb[(size_t)(64 + c) * HWp + p] * fac[128 + b * 64 + c];
}

__global__ void final_kernel(const float* __restrict__ y, const float* __restrict__ zr,
                             const float* __restrict__ zi, const float* __restrict__ gamma,
                             float* __restrict__ out) {
  int i = blockIdx.x * 256 + threadIdx.x;
  if (i >= Bn * Cn * HWp) return;
  int c = (i / HWp) % Cn;
  float a = zr[i], b2 = zi[i];
  out[i] = y[i] + sqrtf(a * a + b2 * b2) * gamma[c];
}

// ---------------------------------------------------------------------------
// f16 WMMA GEMM: D[M,N](f32) = A[M,K](f16 row-major) @ BT (f16 *N-major*:
// BT[n*K + k]) + bias. Each fragment half is 16 contiguous halfs -> b128 loads.
// Block 256 thr = 8 waves, tile 64(M)x32(N); batched on blockIdx.z.
// ---------------------------------------------------------------------------
__global__ __launch_bounds__(256) void gemm_f16_kernel(
    const _Float16* __restrict__ A, const _Float16* __restrict__ BT,
    float* __restrict__ D, const float* __restrict__ bias,
    int M, int N, int K, long long sB, long long sD) {
  const _Float16* Bb = BT + (size_t)blockIdx.z * sB;
  float* Db = D + (size_t)blockIdx.z * sD;
  int tid = threadIdx.x, wave = tid >> 5, lane = tid & 31;
  int wm = wave >> 1, wn = wave & 1;
  int m0 = blockIdx.y * 64 + wm * 16;
  int n0 = blockIdx.x * 32 + wn * 16;
  if (m0 >= M) return;
  int hi = lane >> 4;
  const _Float16* ap = A + (size_t)(m0 + (lane & 15)) * K;        // row of A
  const _Float16* bp = Bb + (size_t)(n0 + (lane & 15)) * K;       // column of B (row of BT)
  int nB = n0 + (lane & 15);
  v8f acc = {};
  for (int k0 = 0; k0 < K; k0 += 32) {
    v8h a0 = *(const v8h*)(ap + k0 + hi * 16);
    v8h a1 = *(const v8h*)(ap + k0 + hi * 16 + 8);
    v16h a = __builtin_shufflevector(a0, a1, 0, 1, 2, 3, 4, 5, 6, 7,
                                     8, 9, 10, 11, 12, 13, 14, 15);
    v8h b0 = *(const v8h*)(bp + k0 + hi * 16);
    v8h b1 = *(const v8h*)(bp + k0 + hi * 16 + 8);
    v16h bf = __builtin_shufflevector(b0, b1, 0, 1, 2, 3, 4, 5, 6, 7,
                                      8, 9, 10, 11, 12, 13, 14, 15);
    acc = __builtin_amdgcn_wmma_f32_16x16x32_f16(false, a, false, bf, (short)0, acc, false, false);
  }
#pragma unroll
  for (int i2 = 0; i2 < 8; ++i2) {
    int m = m0 + hi * 8 + i2;
    float v = acc[i2];
    if (bias) v += bias[m];
    Db[(size_t)m * N + nB] = v;
  }
}

// ---------------------------------------------------------------------------
// f32 WMMA GEMM (v_wmma_f32_16x16x4_f32), DFT matmuls. MODE: 0 D=, 1 D+=, 2 D-=
// ---------------------------------------------------------------------------
template <int MODE>
__global__ __launch_bounds__(256) void gemm_f32_kernel(
    const float* __restrict__ A, int lda, long long sA,
    const float* __restrict__ Bm, int ldb, long long sB,
    float* __restrict__ D, int ldd, long long sD,
    int M, int N, int K) {
  const float* Ab = A + (size_t)blockIdx.z * sA;
  const float* Bb = Bm + (size_t)blockIdx.z * sB;
  float* Db = D + (size_t)blockIdx.z * sD;
  int tid = threadIdx.x, wave = tid >> 5, lane = tid & 31;
  int wm = wave >> 1, wn = wave & 1;
  int m0 = blockIdx.y * 64 + wm * 16;
  int n0 = blockIdx.x * 32 + wn * 16;
  if (m0 >= M || n0 >= N) return;
  int hi = lane >> 4;
  int mA = m0 + (lane & 15);
  int nB = n0 + (lane & 15);
  v8f acc = {};
  for (int k0 = 0; k0 < K; k0 += 4) {
    v2f a = *(const v2f*)(Ab + (size_t)mA * lda + k0 + hi * 2);
    v2f b;
    b[0] = Bb[(size_t)(k0 + hi * 2) * ldb + nB];
    b[1] = Bb[(size_t)(k0 + hi * 2 + 1) * ldb + nB];
    acc = __builtin_amdgcn_wmma_f32_16x16x4_f32(false, a, false, b, (short)0, acc, false, false);
  }
#pragma unroll
  for (int i2 = 0; i2 < 8; ++i2) {
    int m = m0 + hi * 8 + i2;
    float* dp = &Db[(size_t)m * ldd + nB];
    if (MODE == 0) *dp = acc[i2];
    else if (MODE == 1) *dp += acc[i2];
    else *dp -= acc[i2];
  }
}

// ---------------------------------------------------------------------------
// Fused KBA. Per 16-pixel tile:
//  - att tile (1 KB, contiguous) staged to LDS by GLOBAL_LOAD_ASYNC_TO_LDS_B128
//    (ASYNCcnt + s_wait_asynccnt), overlapped with scalar uf-patch staging
//  - per 16-col slab of kb_w (pre-transposed -> b128 loads): one
//    v_wmma_f32_16x16x32_f16 (K = NSET = 32), slab -> wave-private LDS,
//    s_wait_dscnt 0, then contraction against the uf patch.
// Never materializes attk (saves ~470 MB of HBM traffic).
// ---------------------------------------------------------------------------
__global__ __launch_bounds__(256) void kba_kernel(
    const float* __restrict__ uf, const _Float16* __restrict__ attT,
    const _Float16* __restrict__ kbwT, const float* __restrict__ kbb,
    const float* __restrict__ ga1, float* __restrict__ xk) {
  __shared__ _Float16 att_s[16][32];     // 1 KB
  __shared__ float ufp[64][3][18];       // 13.5 KB
  __shared__ float tile_s[8][16][16];    // 8 KB, wave-private slabs
  int tileId = blockIdx.x;
  int b = tileId / (HWp / 16);
  int p0 = (tileId % (HWp / 16)) * 16;   // 16 | 160: tile never crosses a row
  int h = p0 / Wd, w0 = p0 % Wd;
  int tid = threadIdx.x;
  int wave = tid >> 5, lane = tid & 31;

  // async DMA: 64 x 16B chunks of the contiguous att tile -> LDS
  if (wave == 0) {
#pragma unroll
    for (int q = 0; q < 2; ++q) {
      int chunk = q * 32 + lane;
      unsigned long long ga =
          (unsigned long long)(uintptr_t)(attT + (size_t)(b * HWp + p0) * 32 + chunk * 8);
      unsigned int la = (unsigned int)(uintptr_t)(&att_s[0][0]) + chunk * 16;
      asm volatile("global_load_async_to_lds_b128 %0, %1, off"
                   :: "v"(la), "v"(ga) : "memory");
    }
  }
  // overlap: scalar staging of the 3x18 uf patch (64 ch) with bounds/zero pad
  for (int i = tid; i < 64 * 3 * 18; i += 256) {
    int c = i / 54, r = (i % 54) / 18, col = i % 18;
    int hh = h + r - 1, ww = w0 + col - 1;
    float v = 0.f;
    if (hh >= 0 && hh < Hd && ww >= 0 && ww < Wd)
      v = uf[(size_t)(b * Cn + c) * HWp + hh * Wd + ww];
    ufp[c][r][col] = v;
  }
  if (wave == 0) asm volatile("s_wait_asynccnt 0" ::: "memory");
  __syncthreads();

  int hi = lane >> 4, lo = lane & 15;
  v8h a0 = *(const v8h*)(&att_s[lo][hi * 16]);
  v8h a1 = *(const v8h*)(&att_s[lo][hi * 16 + 8]);
  v16h a = __builtin_shufflevector(a0, a1, 0, 1, 2, 3, 4, 5, 6, 7,
                                   8, 9, 10, 11, 12, 13, 14, 15);
  for (int gi = 0; gi < 2; ++gi) {
    int g = wave * 2 + gi;
    float acc0 = 0.f, acc1 = 0.f;  // two (pixel, o) tasks per lane
    for (int t = 0; t < 9; ++t) {
      int ncol = g * 144 + t * 16 + lo;
      const _Float16* kp = kbwT + (size_t)ncol * 32 + hi * 16;
      v8h b0 = *(const v8h*)(kp);
      v8h b1 = *(const v8h*)(kp + 8);
      v16h bf = __builtin_shufflevector(b0, b1, 0, 1, 2, 3, 4, 5, 6, 7,
                                        8, 9, 10, 11, 12, 13, 14, 15);
      v8f c = {};
      c = __builtin_amdgcn_wmma_f32_16x16x32_f16(false, a, false, bf, (short)0, c, false, false);
#pragma unroll
      for (int i2 = 0; i2 < 8; ++i2)
        tile_s[wave][hi * 8 + i2][lo] = c[i2];
      // wave-private slab; LDS ops in-order per wave: wait DS then consume
      asm volatile("s_wait_dscnt 0" ::: "memory");
#pragma unroll 1
      for (int u = 0; u < 2; ++u) {
        int task = lane + u * 32;
        int p = task >> 2, o = task & 3;
        float sum = u ? acc1 : acc0;
        for (int j = 0; j < 16; ++j) {
          int colg = t * 16 + j;
          int oo = colg / 36;
          if (oo == o) {
            int d = colg - oo * 36;
            int ci = d / 9, k2 = d % 9;
            sum += tile_s[wave][p][j] * ufp[g * 4 + ci][k2 / 3][p + (k2 % 3)];
          }
        }
        if (u) acc1 = sum; else acc0 = sum;
      }
    }
#pragma unroll 1
    for (int u = 0; u < 2; ++u) {
      int task = lane + u * 32;
      int p = task >> 2, o = task & 3;
      int c = g * 4 + o;
      float bias = 0.f;
      for (int s = 0; s < 32; ++s)
        bias += (float)att_s[p][s] * kbb[s * Cn + c];
      float acc = u ? acc1 : acc0;
      xk[(size_t)(b * Cn + c) * HWp + p0 + p] = (acc + bias) * ga1[c] + ufp[c][1][p + 1];
    }
  }
}

// ---------------------------------------------------------------------------
// Host orchestration
// ---------------------------------------------------------------------------
extern "C" void kernel_launch(void* const* d_in, const int* in_sizes, int n_in,
                              void* d_out, int out_size, void* d_ws, size_t ws_size,
                              hipStream_t stream) {
  (void)in_sizes; (void)n_in; (void)out_size; (void)ws_size;
  const float* inp   = (const float*)d_in[0];
  const float* n1w   = (const float*)d_in[1];
  const float* n1b   = (const float*)d_in[2];
  const float* n2w   = (const float*)d_in[3];
  const float* n2b   = (const float*)d_in[4];
  const float* scaw  = (const float*)d_in[5];
  const float* scab  = (const float*)d_in[6];
  const float* c11aw = (const float*)d_in[7];
  const float* c11ab = (const float*)d_in[8];
  const float* c11bw = (const float*)d_in[9];
  const float* c11bb = (const float*)d_in[10];
  const float* c1w   = (const float*)d_in[11];
  const float* c1b   = (const float*)d_in[12];
  const float* c21w  = (const float*)d_in[13];
  const float* c21b  = (const float*)d_in[14];
  const float* c2aw  = (const float*)d_in[15];
  const float* c2ab  = (const float*)d_in[16];
  const float* c2cw  = (const float*)d_in[17];
  const float* c2cb  = (const float*)d_in[18];
  const float* c211w = (const float*)d_in[19];
  const float* c211b = (const float*)d_in[20];
  const float* c3w   = (const float*)d_in[21];
  const float* c3b   = (const float*)d_in[22];
  const float* kbw   = (const float*)d_in[23];
  const float* kbb   = (const float*)d_in[24];
  const float* ga1   = (const float*)d_in[25];
  const float* attg  = (const float*)d_in[26];
  const float* beta  = (const float*)d_in[27];
  const float* gamma = (const float*)d_in[28];
  const float* fc1w  = (const float*)d_in[29];
  const float* fc1b  = (const float*)d_in[30];
  const float* fc2w  = (const float*)d_in[31];
  const float* fc2b  = (const float*)d_in[32];
  const float* fscaw = (const float*)d_in[33];
  const float* fscab = (const float*)d_in[34];

  const size_t F = (size_t)Bn * Cn * HWp;  // 3,276,800
  char* ws = (char*)d_ws;
  size_t off = 0;
  auto allocF = [&](size_t n) { float* p = (float*)(ws + off); off += ((n * 4 + 255) & ~(size_t)255); return p; };
  auto allocH = [&](size_t n) { _Float16* p = (_Float16*)(ws + off); off += ((n * 2 + 255) & ~(size_t)255); return p; };

  float* f_xln  = allocF(F);
  float* f_tmp  = allocF(F);
  float* f_x1   = allocF(F);
  float* f_uf   = allocF(F);
  float* f_xk   = allocF(F);
  float* f_y    = allocF(F);
  float* f_t32  = allocF(F / 2);
  float* f_t16  = allocF(F / 4);
  float* f_c211 = allocF(F / 2);
  float* f_rowR = allocF(F);
  float* f_rowI = allocF(F);
  float* f_cf   = allocF(2 * F);
  float* f_fc1  = allocF(4 * F);
  float* f_g2   = allocF(2 * F);
  float* f_fc2  = allocF(2 * F);
  float* f_Fr   = allocF(HWp);
  float* f_Fi   = allocF(HWp);
  float* f_mean = allocF(256);
  float* f_sca  = allocF(128);
  float* f_fac  = allocF(256);

  _Float16* h_xln  = allocH(F);                       // [B][HW][64]  (N-major)
  _Float16* h_cf   = allocH(2 * F);                   // [B][HW][128] (N-major)
  _Float16* h_g2   = allocH(2 * F);                   // [B][HW][128] (N-major)
  _Float16* h_att  = allocH((size_t)Bn * HWp * 32);   // [B][HW][32]
  _Float16* h_kbw  = allocH(32 * 2304);               // [2304][32] transposed
  _Float16* h_wA   = allocH(64 * 64);
  _Float16* h_wB   = allocH(64 * 64);
  _Float16* h_wC   = allocH(64 * 64);
  _Float16* h_w211 = allocH(32 * 64);
  _Float16* h_fc1  = allocH(256 * 128);
  _Float16* h_fc2  = allocH(128 * 128);

  // deterministic buffer reuse (lifetimes disjoint):
  float* f_yln = f_x1;
  float* f_ir  = f_fc1;
  float* f_ii  = f_fc1 + F;
  float* f_jr  = f_rowR;
  float* f_ji  = f_rowI;
  float* f_zr  = f_cf;
  float* f_zi  = f_cf + F;

  auto gemm16 = [&](const _Float16* A, const _Float16* BT, float* D, const float* bias,
                    int M, int N, int K, long long sBm, long long sD, int batches) {
    dim3 grid(N / 32, (M + 63) / 64, batches);
    gemm_f16_kernel<<<grid, 256, 0, stream>>>(A, BT, D, bias, M, N, K, sBm, sD);
  };
  auto gemm32 = [&](const float* A, int lda, long long sA, const float* Bm, int ldb, long long sBm,
                    float* D, int ldd, long long sD, int M, int N, int K, int batches, int mode) {
    dim3 grid((N + 31) / 32, (M + 63) / 64, batches);
    if (mode == 0)
      gemm_f32_kernel<0><<<grid, 256, 0, stream>>>(A, lda, sA, Bm, ldb, sBm, D, ldd, sD, M, N, K);
    else if (mode == 1)
      gemm_f32_kernel<1><<<grid, 256, 0, stream>>>(A, lda, sA, Bm, ldb, sBm, D, ldd, sD, M, N, K);
    else
      gemm_f32_kernel<2><<<grid, 256, 0, stream>>>(A, lda, sA, Bm, ldb, sBm, D, ldd, sD, M, N, K);
  };
  auto pack = [&](const float* in, _Float16* out, size_t n) {
    pack_f16_kernel<<<(int)((n + 255) / 256), 256, 0, stream>>>(in, out, (int)n);
  };
  auto packT = [&](const float* in, _Float16* out, int Cc) {
    packT_f16_kernel<<<(Bn * HWp + 255) / 256, 256, 0, stream>>>(in, out, Cc);
  };

  int pwPix = (Bn * HWp + 255) / 256;
  int pwF   = (int)((F + 255) / 256);
  const long long sAct64  = (long long)HWp * 64;   // per-batch stride of [HW][64] operand
  const long long sAct128 = (long long)HWp * 128;
  const long long CS      = (long long)Cn * HWp;

  init_dft_kernel<<<(Hd * Hd + 255) / 256, 256, 0, stream>>>(f_Fr, f_Fi);
  pack(c11aw, h_wA, 64 * 64);
  pack(c1w,   h_wB, 64 * 64);
  pack(c3w,   h_wC, 64 * 64);
  pack(c211w, h_w211, 32 * 64);
  pack(fc1w,  h_fc1, 256 * 128);
  pack(fc2w,  h_fc2, 128 * 128);
  packT_kbw_kernel<<<(2304 * 32 + 255) / 256, 256, 0, stream>>>(kbw, h_kbw);

  // x = LN1(inp)
  ln2d_kernel<<<pwPix, 256, 0, stream>>>(inp, n1w, n1b, f_xln);
  packT(f_xln, h_xln, 64);

  // sca = conv1x1(mean_hw(x))
  mean_hw_kernel<<<Bn * Cn, 256, 0, stream>>>(f_xln, f_mean);
  small_fc_kernel<<<1, 128, 0, stream>>>(f_mean, 64, scaw, scab, f_sca, 64, 64, 0);

  // x1 = gconv5x5(conv1x1(x))
  gemm16(h_wA, h_xln, f_tmp, c11ab, 64, HWp, 64, sAct64, CS, Bn);
  gconv5x5_kernel<<<pwF, 256, 0, stream>>>(f_tmp, c11bw, c11bb, f_x1);

  // att = c2c(gate(gconv3x3_g32(x)))*attgamma + c211(x) -> transposed f16
  gconv3x3_g32_kernel<<<(int)((F / 2 + 255) / 256), 256, 0, stream>>>(f_xln, c2aw, c2ab, f_t32);
  gate_kernel<<<(int)((F / 4 + 255) / 256), 256, 0, stream>>>(f_t32, f_t16, 16);
  gemm16(h_w211, h_xln, f_c211, c211b, 32, HWp, 64, sAct64, (long long)32 * HWp, Bn);
  att_combine_kernel<<<(int)((F / 2 + 255) / 256), 256, 0, stream>>>(f_t16, c2cw, c2cb, attg,
                                                                     f_c211, h_att);

  // uf = dwconv3x3(conv1x1(x))
  gemm16(h_wB, h_xln, f_tmp, c1b, 64, HWp, 64, sAct64, CS, Bn);
  dwconv3x3_kernel<<<pwF, 256, 0, stream>>>(f_tmp, c21w, c21b, f_uf);

  // xk = KBA(uf, att)*ga1 + uf  (fused WMMA + async-LDS kernel)
  kba_kernel<<<Bn * HWp / 16, 256, 0, stream>>>(f_uf, h_att, h_kbw, kbb, ga1, f_xk);

  // y = inp + conv1x1(xk*x1*sca)*beta
  mul3_kernel<<<pwF, 256, 0, stream>>>(f_xk, f_x1, f_sca, f_xln);
  packT(f_xln, h_xln, 64);
  gemm16(h_wC, h_xln, f_tmp, c3b, 64, HWp, 64, sAct64, CS, Bn);
  residual_kernel<<<pwF, 256, 0, stream>>>(inp, f_tmp, beta, f_y);

  // FCA: fft2(LN2(y)) as DFT matmuls (f32 WMMA)
  ln2d_kernel<<<pwPix, 256, 0, stream>>>(f_y, n2w, n2b, f_yln);
  for (int b = 0; b < Bn; ++b) {
    const float* Xb = f_yln + (size_t)b * Cn * HWp;
    float* Rr = f_rowR + (size_t)b * Cn * HWp;
    float* Ri = f_rowI + (size_t)b * Cn * HWp;
    gemm32(Xb, Wd, HWp, f_Fr, Hd, 0, Rr, Wd, HWp, Hd, Wd, Wd, 64, 0);  // Yr = X@Fr
    gemm32(Xb, Wd, HWp, f_Fi, Hd, 0, Ri, Wd, HWp, Hd, Wd, Wd, 64, 0);  // Yi = X@Fi
    float* CfR = f_cf + (size_t)b * 2 * Cn * HWp;
    float* CfI = CfR + (size_t)Cn * HWp;
    gemm32(f_Fr, Hd, 0, Rr, Wd, HWp, CfR, Wd, HWp, Hd, Wd, Hd, 64, 0); // Zr  = Fr@Yr
    gemm32(f_Fi, Hd, 0, Ri, Wd, HWp, CfR, Wd, HWp, Hd, Wd, Hd, 64, 2); // Zr -= Fi@Yi
    gemm32(f_Fr, Hd, 0, Ri, Wd, HWp, CfI, Wd, HWp, Hd, Wd, Hd, 64, 0); // Zi  = Fr@Yi
    gemm32(f_Fi, Hd, 0, Rr, Wd, HWp, CfI, Wd, HWp, Hd, Wd, Hd, 64, 1); // Zi += Fi@Yr
  }
  // t = fc2(gate(fc1(cf)))
  packT(f_cf, h_cf, 128);
  gemm16(h_fc1, h_cf, f_fc1, fc1b, 256, HWp, 128, sAct128, (long long)256 * HWp, Bn);
  gate_kernel<<<(int)((2 * F + 255) / 256), 256, 0, stream>>>(f_fc1, f_g2, 128);
  packT(f_g2, h_g2, 128);
  gemm16(h_fc2, h_g2, f_fc2, fc2b, 128, HWp, 128, sAct128, (long long)128 * HWp, Bn);
  // fn = (r*(1+ra), im*(1+ia))
  mean_hw_kernel<<<Bn * 128, 256, 0, stream>>>(f_fc2, f_mean);
  small_fc_kernel<<<1, 128, 0, stream>>>(f_mean, 128, fscaw, fscab, f_fac, 64, 64, 1);
  small_fc_kernel<<<1, 128, 0, stream>>>(f_mean + 64, 128, fscaw, fscab, f_fac + 128, 64, 64, 1);
  scale_fn_kernel<<<pwF, 256, 0, stream>>>(f_fc2, f_fac, f_ir, f_ii);
  // ifft2 = conj-DFT matmuls, then |.|
  for (int b = 0; b < Bn; ++b) {
    const float* Ir = f_ir + (size_t)b * Cn * HWp;
    const float* Ii = f_ii + (size_t)b * Cn * HWp;
    float* Jr = f_jr + (size_t)b * Cn * HWp;
    float* Ji = f_ji + (size_t)b * Cn * HWp;
    gemm32(Ir, Wd, HWp, f_Fr, Hd, 0, Jr, Wd, HWp, Hd, Wd, Wd, 64, 0);  // Rr  = Ir@Fr
    gemm32(Ii, Wd, HWp, f_Fi, Hd, 0, Jr, Wd, HWp, Hd, Wd, Wd, 64, 1);  // Rr += Ii@Fi
    gemm32(Ii, Wd, HWp, f_Fr, Hd, 0, Ji, Wd, HWp, Hd, Wd, Wd, 64, 0);  // Ri  = Ii@Fr
    gemm32(Ir, Wd, HWp, f_Fi, Hd, 0, Ji, Wd, HWp, Hd, Wd, Wd, 64, 2);  // Ri -= Ir@Fi
    float* Zr = f_zr + (size_t)b * Cn * HWp;
    float* Zi = f_zi + (size_t)b * Cn * HWp;
    gemm32(f_Fr, Hd, 0, Jr, Wd, HWp, Zr, Wd, HWp, Hd, Wd, Hd, 64, 0);  // Zr  = Fr@Rr
    gemm32(f_Fi, Hd, 0, Ji, Wd, HWp, Zr, Wd, HWp, Hd, Wd, Hd, 64, 1);  // Zr += Fi@Ri
    gemm32(f_Fr, Hd, 0, Ji, Wd, HWp, Zi, Wd, HWp, Hd, Wd, Hd, 64, 0);  // Zi  = Fr@Ri
    gemm32(f_Fi, Hd, 0, Jr, Wd, HWp, Zi, Wd, HWp, Hd, Wd, Hd, 64, 2);  // Zi -= Fi@Rr
  }
  final_kernel<<<pwF, 256, 0, stream>>>(f_y, f_zr, f_zi, gamma, (float*)d_out);
}
